// Detection_54623394071007
// MI455X (gfx1250) — matrix operand order repaired and checked by reading, not compiled, not run on hardware
//
#include <hip/hip_runtime.h>

// ---------------------------------------------------------------------------
// Detection head for gfx1250 (MI455X):
//   roi_pool_bf16 : (1,37,37,512) fmap + 300 boxes -> A_bf16 [320 x 25088]
//   gemm1_wmma    : A(320x25088) @ W1(25088x4069) + b1, ReLU -> H_bf16 [320x4096]
//   gemm2_wmma    : H(304x4096) @ [Wc|Wr](4069x101) + bias -> out (cls|reg)
// Precision: bf16 WMMA (v_wmma_f32_16x16x32_bf16), f32 accumulate.
// ---------------------------------------------------------------------------

typedef __attribute__((ext_vector_type(16))) __bf16       v16bf;
typedef __attribute__((ext_vector_type(8)))  float        v8f;
typedef __attribute__((ext_vector_type(8)))  unsigned int uint8v;
typedef int v4i __attribute__((vector_size(16)));
typedef __attribute__((address_space(1))) v4i* g_v4i_ptr;   // global b128
typedef __attribute__((address_space(3))) v4i* l_v4i_ptr;   // LDS b128

#define K1    25088   // 7*7*512, multiple of 32
#define N1    4069    // hidden
#define N1PAD 4096
#define MPAD  320     // 300 rows padded to 20 tiles of 16
#define KT1   (K1/32) // 784
#define NOUT  101     // 21 cls + 80 reg

#if defined(__gfx1250__) && __has_builtin(__builtin_amdgcn_global_load_async_to_lds_b128)
#define USE_ASYNC_LDS 1
#else
#define USE_ASYNC_LDS 0
#endif

__device__ __forceinline__ void wait_async0() {
#if USE_ASYNC_LDS
#if __has_builtin(__builtin_amdgcn_s_wait_asynccnt)
  __builtin_amdgcn_s_wait_asynccnt(0);
#else
  asm volatile("s_wait_asynccnt 0x0" ::: "memory");
#endif
#endif
}

__device__ __forceinline__ unsigned short f2bf(float f) {
  union { float f; unsigned int u; } x; x.f = f;
  unsigned int u = x.u;
  u += 0x7fffu + ((u >> 16) & 1u);   // round-to-nearest-even
  return (unsigned short)(u >> 16);
}

__device__ __forceinline__ v16bf frag_from(uint4 lo, uint4 hi) {
  uint8v pk;
  pk[0]=lo.x; pk[1]=lo.y; pk[2]=lo.z; pk[3]=lo.w;
  pk[4]=hi.x; pk[5]=hi.y; pk[6]=hi.z; pk[7]=hi.w;
  return __builtin_bit_cast(v16bf, pk);
}

// ---------------------------------------------------------------------------
// Kernel 1: ROI bilinear pooling -> bf16 activation matrix A [MPAD x K1]
// ---------------------------------------------------------------------------
__global__ void __launch_bounds__(256)
roi_pool_bf16(const float* __restrict__ fmap,
              const float* __restrict__ prop,
              unsigned short* __restrict__ A) {
  const int p   = blockIdx.x;
  const int tid = threadIdx.x;
  unsigned short* row = A + (size_t)p * K1;
  if (p >= 300) {                       // zero-pad rows 300..319
    for (int e = tid; e < K1; e += 256) row[e] = 0;
    return;
  }
  const float y1 = prop[p*4+0], x1 = prop[p*4+1];
  const float y2 = prop[p*4+2], x2 = prop[p*4+3];
  for (int e = tid; e < K1; e += 256) {
    int ph = e / 3584;
    int r  = e - ph * 3584;
    int pw = r / 512;
    int c  = r - pw * 512;
    float ty = (float)ph * (1.0f / 6.0f);
    float tx = (float)pw * (1.0f / 6.0f);
    float ys = fminf(fmaxf((y1 + ty * (y2 - y1)) * 36.0f, 0.0f), 36.0f);
    float xs = fminf(fmaxf((x1 + tx * (x2 - x1)) * 36.0f, 0.0f), 36.0f);
    int y0 = (int)floorf(ys); y0 = y0 < 0 ? 0 : (y0 > 35 ? 35 : y0);
    int x0 = (int)floorf(xs); x0 = x0 < 0 ? 0 : (x0 > 35 ? 35 : x0);
    float yf = ys - (float)y0;
    float xf = xs - (float)x0;
    const float* b = fmap + ((y0 * 37 + x0) * 512 + c);
    float tl = b[0], tr = b[512];
    float bl = b[37 * 512], br_ = b[37 * 512 + 512];
    float top = tl + (tr  - tl) * xf;
    float bot = bl + (br_ - bl) * xf;
    row[e] = f2bf(top + (bot - top) * yf);
  }
}

// ---------------------------------------------------------------------------
// Kernel 2: h = relu(A @ W1 + b1), stored bf16 into H [MPAD x N1PAD]
// Grid over N only (64 blocks x 64 cols) so W1 (408 MB) is read exactly once.
// Double-buffered LDS; A tile staged via async global->LDS when available.
// Bs is kept column-major so B fragments load as 2x ds_load_b128 per lane.
// ---------------------------------------------------------------------------
__global__ void __launch_bounds__(256)
gemm1_wmma(const unsigned short* __restrict__ A,
           const float* __restrict__ W1,
           const float* __restrict__ b1,
           unsigned short* __restrict__ H) {
  __shared__ __align__(16) unsigned short As[2][MPAD * 32];  // [m][k] 2x20KB
  __shared__ __align__(16) unsigned short Bs[2][64 * 32];    // [n][k] 2x4KB

  const int tid  = threadIdx.x;
  const int lane = tid & 31;
  const int w    = tid >> 5;          // wave 0..7
  const int nb   = blockIdx.x * 64;   // global N base of block
  const int ncol = (w & 3) * 16;      // wave's N tile inside block
  const int mgrp = (w >> 2) * 10;     // wave's first M tile (0 or 10)

  // B staging map: 64 cols x 4 k-groups of 8 (coalesced across lanes)
  const int  bscol = tid & 63;
  const int  bkgrp = (tid >> 6) * 8;
  const bool bok   = (nb + bscol) < N1;

  // fragment index maps (ISA 16-bit 16x32 layout)
  const int mrow  = lane & 15;
  const int aoff  = (lane < 16) ? 0 : 8;    // A k sub-offset
  const int koff  = (lane < 16) ? 0 : 8;    // B k sub-offset
  const int bcolf = ncol + (lane & 15);     // B frag column in block

  v8f zero = {0.f, 0.f, 0.f, 0.f, 0.f, 0.f, 0.f, 0.f};
  v8f acc[10];
#pragma unroll
  for (int t = 0; t < 10; ++t) acc[t] = zero;

  // --- staging helpers -----------------------------------------------------
  auto issue_a = [&](int kbase, int buf) {
#pragma unroll
    for (int i = 0; i < 5; ++i) {
      int q    = tid + i * 256;
      int row  = q >> 2;
      int part = (q & 3) * 8;
#if USE_ASYNC_LDS
      __builtin_amdgcn_global_load_async_to_lds_b128(
          (g_v4i_ptr)(A + (size_t)row * K1 + kbase + part),
          (l_v4i_ptr)(&As[buf][row * 32 + part]),
          0, 0);
#else
      uint4 v = *(const uint4*)(A + (size_t)row * K1 + kbase + part);
      *(uint4*)(&As[buf][row * 32 + part]) = v;
#endif
    }
  };
  auto load_b = [&](int kbase, float* r) {
#pragma unroll
    for (int j = 0; j < 8; ++j)
      r[j] = bok ? W1[(size_t)(kbase + bkgrp + j) * N1 + nb + bscol] : 0.0f;
  };
  auto store_b = [&](int buf, const float* r) {
    uint4 p;
    p.x = (unsigned)f2bf(r[0]) | ((unsigned)f2bf(r[1]) << 16);
    p.y = (unsigned)f2bf(r[2]) | ((unsigned)f2bf(r[3]) << 16);
    p.z = (unsigned)f2bf(r[4]) | ((unsigned)f2bf(r[5]) << 16);
    p.w = (unsigned)f2bf(r[6]) | ((unsigned)f2bf(r[7]) << 16);
    *(uint4*)(&Bs[buf][bscol * 32 + bkgrp]) = p;
  };

  // --- prologue: stage tile 0 into buffer 0 --------------------------------
  float breg[8];
  issue_a(0, 0);
  load_b(0, breg);
  store_b(0, breg);
  wait_async0();
  __syncthreads();

  // --- main K loop (double-buffered) ---------------------------------------
  for (int kt = 0; kt < KT1; ++kt) {
    const int cur  = kt & 1;
    const int nxt  = cur ^ 1;
    const bool more = (kt + 1) < KT1;

    if (more) {                       // issue next tile while computing
      const int knext = (kt + 1) * 32;
      issue_a(knext, nxt);
      load_b(knext, breg);
      __builtin_prefetch(W1 + (size_t)(knext + 32 + bkgrp) * N1 + nb + bscol,
                         0, 1);       // pull tile kt+2 toward L2
    }

    // B fragment: 2x ds_load_b128 (Bs is [n][k] col-major)
    uint4 blo = *(const uint4*)(&Bs[cur][bcolf * 32 + koff]);
    uint4 bhi = *(const uint4*)(&Bs[cur][bcolf * 32 + koff + 16]);
    v16bf bfrag = frag_from(blo, bhi);

#pragma unroll
    for (int t = 0; t < 10; ++t) {
      int arow = (mgrp + t) * 16 + mrow;
      uint4 alo = *(const uint4*)(&As[cur][arow * 32 + aoff]);
      uint4 ahi = *(const uint4*)(&As[cur][arow * 32 + 16 + aoff]);
      v16bf afrag = frag_from(alo, ahi);
      acc[t] = __builtin_amdgcn_wmma_f32_16x16x32_bf16(
                   false, afrag, false, bfrag, (short)0, acc[t], false, false);
    }

    if (more) store_b(nxt, breg);
    wait_async0();
    __syncthreads();
  }

  // --- epilogue: bias + ReLU, store bf16 (zero the padded columns) ---------
  const int   n    = nb + ncol + (lane & 15);
  const float bias = (n < N1) ? b1[n] : 0.0f;
#pragma unroll
  for (int t = 0; t < 10; ++t) {
    int mbase = (mgrp + t) * 16 + ((lane < 16) ? 0 : 8);
#pragma unroll
    for (int v = 0; v < 8; ++v) {
      float val = fmaxf(acc[t][v] + bias, 0.0f);
      H[(size_t)(mbase + v) * N1PAD + n] =
          (n < N1) ? f2bf(val) : (unsigned short)0;
    }
  }
}

// ---------------------------------------------------------------------------
// Kernel 3: [cls|reg] = H @ [Wc|Wr] + [bc|br].  One wave per 16x16 tile.
// ---------------------------------------------------------------------------
__device__ __forceinline__ float head_w(const float* Wc, const float* Wr,
                                        int k, int n) {
  if (k >= N1)  return 0.0f;
  if (n < 21)   return Wc[k * 21 + n];
  if (n < NOUT) return Wr[k * 80 + (n - 21)];
  return 0.0f;
}

__global__ void __launch_bounds__(32)
gemm2_wmma(const unsigned short* __restrict__ H,
           const float* __restrict__ Wc, const float* __restrict__ bc,
           const float* __restrict__ Wr, const float* __restrict__ br,
           float* __restrict__ out) {
  const int lane = threadIdx.x & 31;
  const int m0   = blockIdx.x * 16;
  const int n0   = blockIdx.y * 16;
  const int aoff = (lane < 16) ? 0 : 16;
  const int koff = (lane < 16) ? 0 : 16;
  const int arow = m0 + (lane & 15);
  const int col  = n0 + (lane & 15);

  v8f acc = {0.f, 0.f, 0.f, 0.f, 0.f, 0.f, 0.f, 0.f};

  for (int kt = 0; kt < N1PAD / 32; ++kt) {
    const int kb = kt * 32;
    uint4 alo = *(const uint4*)(H + (size_t)arow * N1PAD + kb + aoff);
    uint4 ahi = *(const uint4*)(H + (size_t)arow * N1PAD + kb + aoff + 8);
    v16bf afrag = frag_from(alo, ahi);

    unsigned int bw[8];
#pragma unroll
    for (int j = 0; j < 8; ++j) {
      int k0 = kb + koff + 2 * j;
      unsigned int e0 = f2bf(head_w(Wc, Wr, k0,     col));
      unsigned int e1 = f2bf(head_w(Wc, Wr, k0 + 1, col));
      bw[j] = e0 | (e1 << 16);
    }
    uint8v pk;
    pk[0]=bw[0]; pk[1]=bw[1]; pk[2]=bw[2]; pk[3]=bw[3];
    pk[4]=bw[4]; pk[5]=bw[5]; pk[6]=bw[6]; pk[7]=bw[7];
    v16bf bfrag = __builtin_bit_cast(v16bf, pk);

    acc = __builtin_amdgcn_wmma_f32_16x16x32_bf16(
              false, afrag, false, bfrag, (short)0, acc, false, false);
  }

  // scatter into concatenated (cls_logits[300x21], reg[300x80]) output
#pragma unroll
  for (int v = 0; v < 8; ++v) {
    int m = m0 + v + ((lane < 16) ? 0 : 8);
    if (m < 300) {
      float val = acc[v];
      if (col < 21)        out[m * 21 + col]               = val + bc[col];
      else if (col < NOUT) out[6300 + m * 80 + (col - 21)] = val + br[col - 21];
    }
  }
}

// ---------------------------------------------------------------------------
extern "C" void kernel_launch(void* const* d_in, const int* in_sizes, int n_in,
                              void* d_out, int out_size, void* d_ws, size_t ws_size,
                              hipStream_t stream) {
  (void)in_sizes; (void)n_in; (void)out_size; (void)ws_size;
  const float* fmap = (const float*)d_in[0];
  const float* prop = (const float*)d_in[1];
  const float* W1   = (const float*)d_in[2];
  const float* b1   = (const float*)d_in[3];
  const float* Wc   = (const float*)d_in[4];
  const float* bc   = (const float*)d_in[5];
  const float* Wr   = (const float*)d_in[6];
  const float* br   = (const float*)d_in[7];
  float* out = (float*)d_out;

  unsigned short* Abf = (unsigned short*)d_ws;              // 320*25088 bf16
  unsigned short* Hbf = Abf + (size_t)MPAD * K1;            // 320*4096 bf16

  roi_pool_bf16<<<MPAD, 256, 0, stream>>>(fmap, prop, Abf);
  gemm1_wmma<<<N1PAD / 64, 256, 0, stream>>>(Abf, W1, b1, Hbf);
  gemm2_wmma<<<dim3(19, 7), 32, 0, stream>>>(Hbf, Wc, bc, Wr, br, out);
}
